// RNN_70437463654737
// MI455X (gfx1250) — compile-verified, matrix operand order checked by reading, not run on hardware
//
#include <hip/hip_runtime.h>
#include <hip/hip_bf16.h>

#define BB 256
#define SS 10000
#define HH 16
#define KK 5
#define TT 2000   // S / K
#define GG 64     // 4*H

typedef __attribute__((ext_vector_type(16))) _Float16 v16h;
typedef __attribute__((ext_vector_type(8)))  _Float16 v8h;
typedef __attribute__((ext_vector_type(8)))  float    v8f;

__device__ __forceinline__ float fast_rcp(float x) {
#if __has_builtin(__builtin_amdgcn_rcpf)
  return __builtin_amdgcn_rcpf(x);
#else
  return 1.0f / x;
#endif
}

__device__ __forceinline__ float fast_rsqrt(float x) {
#if __has_builtin(__builtin_amdgcn_rsqf)
  return __builtin_amdgcn_rsqf(x);
#else
  return rsqrtf(x);
#endif
}

__device__ __forceinline__ float sigmoidf_(float x) {
  return fast_rcp(1.0f + __expf(-x));
}

__device__ __forceinline__ float fast_tanh(float x) {
#if defined(__gfx1250__) && __has_builtin(__builtin_amdgcn_tanhf)
  return __builtin_amdgcn_tanhf(x);    // v_tanh_f32 (CDNA5 TRANS op)
#else
  float e = __expf(2.0f * x);
  return 1.0f - 2.0f * fast_rcp(e + 1.0f);
#endif
}

__device__ __forceinline__ v8f wmma_f16(v16h a, v16h b, v8f c) {
  // 8 args: (neg_a, A, neg_b, B, c_mod, C, reuse_a, reuse_b)
  return __builtin_amdgcn_wmma_f32_16x16x32_f16(false, a, false, b, (short)0, c, false, false);
}

// ---------------------------------------------------------------------------
// Kernel 1: u = relu(x*W_mlp+b_mlp); seq[t,b,:] = sigmoid(conv(u) + b_conv)
// One thread per (t,b). Stride==kernel==5 -> disjoint windows of 5 scalars.
// ---------------------------------------------------------------------------
__global__ void conv_seq_kernel(const float* __restrict__ x,
                                const float* __restrict__ W_mlp,
                                const float* __restrict__ b_mlp,
                                const float* __restrict__ W_conv,
                                const float* __restrict__ b_conv,
                                _Float16* __restrict__ seq)
{
  __shared__ float sWc[HH * HH * KK];
  __shared__ float sWm[HH], sBm[HH], sBc[HH];
  int tid = threadIdx.x;
  for (int i = tid; i < HH * HH * KK; i += blockDim.x) sWc[i] = W_conv[i];
  if (tid < HH) { sWm[tid] = W_mlp[tid]; sBm[tid] = b_mlp[tid]; sBc[tid] = b_conv[tid]; }
  __syncthreads();

  int gid = blockIdx.x * blockDim.x + tid;  // gid = t*B + b
  int t = gid >> 8;                         // B == 256
  int b = gid & (BB - 1);

  float u[KK][HH];
  const float* xp = x + (size_t)b * SS + (size_t)t * KK;
#pragma unroll
  for (int k = 0; k < KK; ++k) {
    float xv = xp[k];
#pragma unroll
    for (int i = 0; i < HH; ++i) {
      float v = xv * sWm[i] + sBm[i];
      u[k][i] = v > 0.f ? v : 0.f;
    }
  }
  _Float16* outp = seq + (size_t)gid * HH;
  for (int o = 0; o < HH; ++o) {
    float acc = sBc[o];
#pragma unroll
    for (int i = 0; i < HH; ++i)
#pragma unroll
      for (int k = 0; k < KK; ++k)
        acc += sWc[(o * HH + i) * KK + k] * u[k][i];
    outp[o] = (_Float16)sigmoidf_(acc);
  }
}

// ---------------------------------------------------------------------------
// Kernel 2: gates_x = LN(seq @ Wx^T)*gx + bx + b_gates   (T*B rows, 64 cols)
// One wave per 16-row M-tile (fixed t, 16 consecutive b). 4 N-tiles of 16.
// K=16 zero-padded into the K=32 f16 WMMA.
// ---------------------------------------------------------------------------
__global__ void gatesx_kernel(const _Float16* __restrict__ seq,
                              const float* __restrict__ Wx,
                              const float* __restrict__ gx,
                              const float* __restrict__ bx,
                              const float* __restrict__ b_gates,
                              float* __restrict__ gates_x)
{
  __shared__ __align__(16) float hh[16][GG];
  __shared__ float smean[16], srstd[16];
  int lane = threadIdx.x;     // 0..31
  int m  = blockIdx.x;        // tile = t*16 + bt
  int t  = m >> 4;
  int bt = m & 15;
  int hs = lane >> 4;         // half-wave select
  int lq = lane & 15;

  // A fragment: 16x(K=16) f16, rows = batch; 16-bit A layout, pad K 16..31 = 0
  v16h a;
  const _Float16* ap = seq + ((size_t)t * BB + bt * 16 + lq) * HH + hs * 8;
#pragma unroll
  for (int e = 0; e < 8; ++e) a[e] = ap[e];
#pragma unroll
  for (int e = 8; e < 16; ++e) a[e] = (_Float16)0.0f;

#pragma unroll
  for (int nt = 0; nt < 4; ++nt) {
    v16h bf;
    const float* wp = Wx + (nt * 16 + lq) * HH + hs * 8;  // B[k][n] = Wx[n][k]
#pragma unroll
    for (int e = 0; e < 8; ++e) bf[e] = (_Float16)wp[e];
#pragma unroll
    for (int e = 8; e < 16; ++e) bf[e] = (_Float16)0.0f;
    v8f acc = {};
    acc = wmma_f16(a, bf, acc);
#pragma unroll
    for (int v = 0; v < 8; ++v)
      hh[v + 8 * hs][nt * 16 + lq] = acc[v];  // C layout: M = v + 8*hs, N = lq
  }
  __syncthreads();

  if (lane < 16) {
    float s = 0.f, s2 = 0.f;
    for (int c = 0; c < GG; ++c) { float v = hh[lane][c]; s += v; s2 += v * v; }
    float mean = s * (1.0f / GG);
    float var  = s2 * (1.0f / GG) - mean * mean;
    smean[lane] = mean;
    srstd[lane] = fast_rsqrt(var + 1e-5f);
  }
  __syncthreads();

  float* op = gates_x + ((size_t)t * BB + bt * 16) * GG;
  for (int idx = lane; idx < 16 * GG; idx += 32) {
    int r = idx >> 6, c = idx & 63;
    op[idx] = (hh[r][c] - smean[r]) * srstd[r] * gx[c] + bx[c] + b_gates[c];
  }
}

// ---------------------------------------------------------------------------
// Kernel 3: persistent LSTM scan. One block (2 waves, 64 thr) per batch-tile
// of 16 rows; 2000 sequential steps, 2 barriers/step.
// Thread (r,q) owns the strided gate columns {16s + 4q + k} so its LN partial
// sums reduce to full row statistics via 4-lane shfl_xor AND it holds all four
// gate activations for its own 4 cells in registers (no exchange stage).
// ---------------------------------------------------------------------------
__global__ void lstm_kernel(const float* __restrict__ gates_x,
                            const float* __restrict__ Wh,
                            const float* __restrict__ gh,
                            const float* __restrict__ bh,
                            const float* __restrict__ gc,
                            const float* __restrict__ bc,
                            const float* __restrict__ W_cls,
                            const float* __restrict__ b_cls,
                            const float* __restrict__ h0,
                            const float* __restrict__ c0,
                            float* __restrict__ out)
{
  __shared__ __align__(16) _Float16 h16[16][HH];  // hidden state (WMMA operand prec)
  __shared__ __align__(16) float    hh[16][GG];   // raw h-gates from WMMA
  int tid  = threadIdx.x;           // 0..63
  int wave = tid >> 5;
  int lane = tid & 31;
  int hs = (lane >> 4) & 1;
  int lq = lane & 15;
  int bt = blockIdx.x;              // 0..15
  int r = tid >> 2;                 // row 0..15 for LN/update stages
  int q = tid & 3;
  int j0 = q * 4;                   // this thread's 4 cells

  // init h (f16)
#pragma unroll
  for (int k = 0; k < 4; ++k) {
    int off = tid * 4 + k;          // 0..255
    int rr = off >> 4, jj = off & 15;
    h16[rr][jj] = (_Float16)h0[(size_t)(bt * 16 + rr) * HH + jj];
  }
  // init c (each thread owns 4 cells of its row)
  float creg[4];
#pragma unroll
  for (int k = 0; k < 4; ++k)
    creg[k] = c0[(size_t)(bt * 16 + r) * HH + j0 + k];

  // Wh B fragments: wave w owns N-tiles 2w and 2w+1 (constant over the scan)
  v16h bw0, bw1;
  {
    const float* w0 = Wh + ((2 * wave + 0) * 16 + lq) * HH + hs * 8;
    const float* w1 = Wh + ((2 * wave + 1) * 16 + lq) * HH + hs * 8;
#pragma unroll
    for (int e = 0; e < 8; ++e) { bw0[e] = (_Float16)w0[e]; bw1[e] = (_Float16)w1[e]; }
#pragma unroll
    for (int e = 8; e < 16; ++e) { bw0[e] = (_Float16)0.0f; bw1[e] = (_Float16)0.0f; }
  }

  // per-thread LN affine constants for owned columns 16s + j0 + k
  float ghv[4][4], bhv[4][4];
#pragma unroll
  for (int s = 0; s < 4; ++s)
#pragma unroll
    for (int k = 0; k < 4; ++k) {
      ghv[s][k] = gh[16 * s + j0 + k];
      bhv[s][k] = bh[16 * s + j0 + k];
    }
  float gcv[4], bcv[4];
#pragma unroll
  for (int k = 0; k < 4; ++k) { gcv[k] = gc[j0 + k]; bcv[k] = bc[j0 + k]; }

  const float* gp = gates_x + (size_t)(bt * 16 + r) * GG;  // this row's x-gates

  __syncthreads();

  for (int t = 0; t < TT; ++t) {
    // ---- hh = h @ Wh^T (f16 operands, f32 accumulate) ----
    v8h ah = *(const v8h*)&h16[lq][hs * 8];    // one 16B LDS load, no converts
    v16h a;
#pragma unroll
    for (int e = 0; e < 8; ++e) a[e] = ah[e];
#pragma unroll
    for (int e = 8; e < 16; ++e) a[e] = (_Float16)0.0f;
    v8f acc0 = {}, acc1 = {};
    acc0 = wmma_f16(a, bw0, acc0);
    acc1 = wmma_f16(a, bw1, acc1);
#pragma unroll
    for (int v = 0; v < 8; ++v) {
      hh[v + 8 * hs][(2 * wave + 0) * 16 + lq] = acc0[v];
      hh[v + 8 * hs][(2 * wave + 1) * 16 + lq] = acc1[v];
    }
    if (t + 1 < TT)  // hide global latency of next step's x-gates
      __builtin_prefetch(gp + (size_t)(t + 1) * BB * GG, 0, 0);
    __syncthreads();

    // ---- LN over 64 gates per row (strided ownership) + gates_x + acts ----
    float raw[4][4];
    float s1 = 0.f, s2 = 0.f;
#pragma unroll
    for (int s = 0; s < 4; ++s) {
      float4 hv = *(const float4*)&hh[r][16 * s + j0];
      raw[s][0] = hv.x; raw[s][1] = hv.y; raw[s][2] = hv.z; raw[s][3] = hv.w;
#pragma unroll
      for (int k = 0; k < 4; ++k) { s1 += raw[s][k]; s2 += raw[s][k] * raw[s][k]; }
    }
    s1 += __shfl_xor(s1, 1, 32);  s1 += __shfl_xor(s1, 2, 32);
    s2 += __shfl_xor(s2, 1, 32);  s2 += __shfl_xor(s2, 2, 32);
    float mean = s1 * (1.0f / GG);
    float var  = s2 * (1.0f / GG) - mean * mean;
    float rstd = fast_rsqrt(var + 1e-5f);

    const float* gxt = gp + (size_t)t * BB * GG;
    float act[4][4];
#pragma unroll
    for (int s = 0; s < 4; ++s) {
      float4 gv = *(const float4*)(gxt + 16 * s + j0);
      float gva[4] = { gv.x, gv.y, gv.z, gv.w };
#pragma unroll
      for (int k = 0; k < 4; ++k) {
        float v = (raw[s][k] - mean) * rstd * ghv[s][k] + bhv[s][k] + gva[k];
        act[s][k] = (s == 2) ? fast_tanh(v) : sigmoidf_(v);
      }
    }

    // ---- c = sig(f)*c + sig(i)*tanh(g);  h = sig(o)*tanh(LN(c)) ----
    // act[0]=i, act[1]=f, act[2]=g, act[3]=o for cells j0..j0+3 (in registers)
    float cn[4];
    float sc = 0.f, sc2 = 0.f;
#pragma unroll
    for (int k = 0; k < 4; ++k) {
      cn[k] = act[1][k] * creg[k] + act[0][k] * act[2][k];
      sc += cn[k]; sc2 += cn[k] * cn[k];
    }
    sc  += __shfl_xor(sc, 1, 32);   sc  += __shfl_xor(sc, 2, 32);
    sc2 += __shfl_xor(sc2, 1, 32);  sc2 += __shfl_xor(sc2, 2, 32);
    float meanc = sc * (1.0f / HH);
    float varc  = sc2 * (1.0f / HH) - meanc * meanc;
    float rstdc = fast_rsqrt(varc + 1e-5f);
#pragma unroll
    for (int k = 0; k < 4; ++k) {
      float cl = (cn[k] - meanc) * rstdc * gcv[k] + bcv[k];
      h16[r][j0 + k] = (_Float16)(act[3][k] * fast_tanh(cl));
      creg[k] = cn[k];
    }
    __syncthreads();
  }

  // ---- classifier on final h ----
  if (tid < 16) {
    float s = b_cls[0];
#pragma unroll
    for (int j = 0; j < HH; ++j) s += (float)h16[tid][j] * W_cls[j];
    out[bt * 16 + tid] = sigmoidf_(s);
  }
}

// ---------------------------------------------------------------------------
extern "C" void kernel_launch(void* const* d_in, const int* in_sizes, int n_in,
                              void* d_out, int out_size, void* d_ws, size_t ws_size,
                              hipStream_t stream)
{
  const float* x       = (const float*)d_in[0];
  const float* W_mlp   = (const float*)d_in[1];
  const float* b_mlp   = (const float*)d_in[2];
  const float* W_conv  = (const float*)d_in[3];
  const float* b_conv  = (const float*)d_in[4];
  const float* Wx      = (const float*)d_in[5];
  const float* Wh      = (const float*)d_in[6];
  const float* b_gates = (const float*)d_in[7];
  const float* gx      = (const float*)d_in[8];
  const float* bx      = (const float*)d_in[9];
  const float* gh      = (const float*)d_in[10];
  const float* bh      = (const float*)d_in[11];
  const float* gc      = (const float*)d_in[12];
  const float* bc      = (const float*)d_in[13];
  const float* W_cls   = (const float*)d_in[14];
  const float* b_cls   = (const float*)d_in[15];
  const float* h0      = (const float*)d_in[16];
  const float* c0      = (const float*)d_in[17];
  float* out = (float*)d_out;

  _Float16* seq = (_Float16*)d_ws;                       // T*B*H f16  (~16.4 MB)
  size_t seq_bytes = (size_t)TT * BB * HH * sizeof(_Float16);
  float* gates_x = (float*)((char*)d_ws + seq_bytes);    // T*B*64 f32 (~131 MB)

  conv_seq_kernel<<<(TT * BB) / 256, 256, 0, stream>>>(x, W_mlp, b_mlp, W_conv, b_conv, seq);
  gatesx_kernel<<<TT * (BB / 16), 32, 0, stream>>>(seq, Wx, gx, bx, b_gates, gates_x);
  lstm_kernel<<<BB / 16, 64, 0, stream>>>(gates_x, Wh, gh, bh, gc, bc,
                                          W_cls, b_cls, h0, c0, out);
}
